// Attention_2070174236874
// MI455X (gfx1250) — matrix operand order repaired and checked by reading, not compiled, not run on hardware
//
#include <hip/hip_runtime.h>

// ---------------------------------------------------------------------------
// MI455X (gfx1250) attention: bf16 WMMA pipeline with flash-attention core.
//   B=2, N=2048, C=1024, H=16, HD=64
// Workspace layout (48 MiB total, assumed <= ws_size):
//   xb[4096*1024] bf16, wqkv[3072*1024] bf16, wproj[1024*1024] bf16,
//   Q[32][2048][64] bf16, K[32][2048][64] bf16, Vt[32][64][2048] bf16,
//   vals[4096*1024] bf16
// ---------------------------------------------------------------------------

typedef __attribute__((ext_vector_type(16))) __bf16 v16bf;
typedef __attribute__((ext_vector_type(8)))  float  v8f;

struct alignas(16) U4 { unsigned int x, y, z, w; };
union FragCast { v16bf v; U4 q[2]; };

__device__ inline unsigned short f2bf(float f) {
  unsigned u = __builtin_bit_cast(unsigned, f);
  u += 0x7FFFu + ((u >> 16) & 1u);          // round-to-nearest-even
  return (unsigned short)(u >> 16);
}

// CDNA5 async copy: global -> LDS, 16B per lane, tracked by ASYNCcnt.
// lds_off = addrspace(3) byte offset (low 32 bits of a generic LDS pointer).
__device__ inline void async_load_b128(unsigned lds_off, const void* g) {
  asm volatile("global_load_async_to_lds_b128 %0, %1, off"
               :: "v"(lds_off), "v"((unsigned long long)(uintptr_t)g)
               : "memory");
}

// A-frag (16x32 bf16, ISA 7.12.2): lane<16 -> row=lane, K 0..7 & 16..23;
// lane>=16 -> row=lane-16, K 8..15 & 24..31.  base is row-major bf16, ld elems.
__device__ inline v16bf load_a_frag(const unsigned short* base, int ld, int lane) {
  const int half = lane >> 4, row = lane & 15;
  const unsigned short* p = base + row * ld + half * 8;
  FragCast f;
  f.q[0] = *reinterpret_cast<const U4*>(p);
  f.q[1] = *reinterpret_cast<const U4*>(p + 16);
  return f.v;
}

// B-frag (32x16 = KxN bf16): lane -> col = lane&15, K = half*8..+7 & 16+half*8..+7
// base is the N-major matrix (row n holds its K values contiguously), ld elems.
__device__ inline v16bf load_b_frag(const unsigned short* base, int ld, int lane) {
  const int half = lane >> 4, col = lane & 15;
  const unsigned short* p = base + col * ld + half * 8;
  FragCast f;
  f.q[0] = *reinterpret_cast<const U4*>(p);
  f.q[1] = *reinterpret_cast<const U4*>(p + 16);
  return f.v;
}

__device__ inline v8f wmma_bf16(v16bf a, v16bf b, v8f c) {
  return __builtin_amdgcn_wmma_f32_16x16x32_bf16(false, a, false, b,
                                                 (short)0, c, false, false);
}

// ---------------------------------------------------------------------------
__global__ void k_cvt_bf16(const float* __restrict__ s,
                           unsigned short* __restrict__ d, int n) {
  int i = blockIdx.x * blockDim.x + threadIdx.x;
  const int stride = gridDim.x * blockDim.x;
  for (; i < n; i += stride) d[i] = f2bf(s[i]);
}

// ---------------------------------------------------------------------------
// out[m, j] = sum_k A[m,k] * W[j,k] + bias[j]
// Tiles staged with gfx1250 async-to-LDS loads (ASYNCcnt), double-buffered:
// tile t+1 is in flight while tile t is consumed by WMMA.
// EPI=0: scatter Q/K (row-major [bh][n][64]) and Vt ([bh][64][n]) as bf16
// EPI=1: f32 dense output [M][Nout]
template <int EPI>
__global__ __launch_bounds__(256) void k_gemm_bf16(
    const unsigned short* __restrict__ A,
    const unsigned short* __restrict__ W,
    const float* __restrict__ bias,
    float* __restrict__ outF,
    unsigned short* __restrict__ Qb,
    unsigned short* __restrict__ Kb,
    unsigned short* __restrict__ Vtb,
    int M, int Nout, int K) {
  __shared__ alignas(16) unsigned short As[2][128][40];  // 80B row stride
  __shared__ alignas(16) unsigned short Bs[2][128][40];
  const int tid = threadIdx.x, lane = tid & 31, wave = tid >> 5;
  const int wm = wave & 3, wn = wave >> 2;               // 4x2 wave grid
  const int m0 = blockIdx.x * 128, n0 = blockIdx.y * 128;

  // this thread's 16B staging chunks: rows r0 / r0+64, col chunk kc0
  const int r0 = tid >> 2, kc0 = (tid & 3) * 8;

  const v8f vzero = {0.f, 0.f, 0.f, 0.f, 0.f, 0.f, 0.f, 0.f};
  v8f acc[2][4];
  for (int i = 0; i < 2; ++i)
    for (int j = 0; j < 4; ++j) acc[i][j] = vzero;

  // issue async loads for one 128x32 A-tile + 128x32 B-tile (4 per thread)
  auto issue_tile = [&](int buf, int kb) {
    async_load_b128((unsigned)(uintptr_t)&As[buf][r0][kc0],
                    A + (size_t)(m0 + r0) * K + kb + kc0);
    async_load_b128((unsigned)(uintptr_t)&As[buf][r0 + 64][kc0],
                    A + (size_t)(m0 + r0 + 64) * K + kb + kc0);
    async_load_b128((unsigned)(uintptr_t)&Bs[buf][r0][kc0],
                    W + (size_t)(n0 + r0) * K + kb + kc0);
    async_load_b128((unsigned)(uintptr_t)&Bs[buf][r0 + 64][kc0],
                    W + (size_t)(n0 + r0 + 64) * K + kb + kc0);
  };

  issue_tile(0, 0);
  const int T = K / 32;
  for (int t = 0; t < T; ++t) {
    const int cur = t & 1;
    if (t + 1 < T) {
      issue_tile(cur ^ 1, (t + 1) * 32);
      // async loads complete in order: <=4 outstanding => tile t has landed
      asm volatile("s_wait_asynccnt 0x4" ::: "memory");
    } else {
      asm volatile("s_wait_asynccnt 0x0" ::: "memory");
    }
    __syncthreads();
    v16bf af0 = load_a_frag(&As[cur][wm * 32 + 0][0], 40, lane);
    v16bf af1 = load_a_frag(&As[cur][wm * 32 + 16][0], 40, lane);
    #pragma unroll
    for (int ni = 0; ni < 4; ++ni) {
      v16bf bf = load_b_frag(&Bs[cur][wn * 64 + ni * 16][0], 40, lane);
      acc[0][ni] = wmma_bf16(af0, bf, acc[0][ni]);
      acc[1][ni] = wmma_bf16(af1, bf, acc[1][ni]);
    }
    __syncthreads();   // protect buf (cur) from being overwritten next iter
  }

  const int rl = lane >> 4, cl = lane & 15;
  #pragma unroll
  for (int mi = 0; mi < 2; ++mi)
    #pragma unroll
    for (int ni = 0; ni < 4; ++ni)
      #pragma unroll
      for (int r = 0; r < 8; ++r) {
        const int row = m0 + wm * 32 + mi * 16 + r + 8 * rl;
        const int col = n0 + wn * 64 + ni * 16 + cl;
        const float v = acc[mi][ni][r] + bias[col];
        if (EPI == 1) {
          outF[(size_t)row * Nout + col] = v;
        } else {
          const int g = col >> 10, h = (col >> 6) & 15, d = col & 63;
          const int bb = row >> 11, n = row & 2047;
          const size_t bh = (size_t)(bb * 16 + h);
          if (g == 0)      Qb[(bh * 2048 + n) * 64 + d] = f2bf(v);
          else if (g == 1) Kb[(bh * 2048 + n) * 64 + d] = f2bf(v);
          else             Vtb[(bh * 64 + d) * 2048 + n] = f2bf(v);
        }
      }
}

// ---------------------------------------------------------------------------
// Flash attention: 4 waves/WG, each wave owns a 16-row query tile; keys
// streamed in tiles of 64.  Row-max via shfl butterfly (only unavoidable
// cross-lane op); row-sums come for free from an extra WMMA against an
// all-ones B matrix (the sum column rescales exactly like the O accum).
// Softmax runs in exp2 domain: a = s * (scale*log2e) + bias.
__global__ __launch_bounds__(128) void k_attn(
    const unsigned short* __restrict__ Qb,
    const unsigned short* __restrict__ Kb,
    const unsigned short* __restrict__ Vtb,
    const int* __restrict__ mask,
    unsigned short* __restrict__ vals) {
  constexpr int N = 2048, HD = 64;
  constexpr int PLD = 88;                       // P row stride: 176B, 16B-aligned
  __shared__ alignas(16) unsigned short Ps[4][16][PLD];
  const int tid = threadIdx.x, lane = tid & 31, wave = tid >> 5;
  const int qblk = blockIdx.x & 31, bh = blockIdx.x >> 5;
  const int b = bh >> 4, h = bh & 15;
  const int m0 = qblk * 64 + wave * 16;

  const unsigned short* Qh = Qb + (size_t)bh * N * HD;
  const unsigned short* Kh = Kb + (size_t)bh * N * HD;
  const unsigned short* Vh = Vtb + (size_t)bh * HD * N;

  const v16bf qf0 = load_a_frag(Qh + (size_t)m0 * HD + 0, HD, lane);
  const v16bf qf1 = load_a_frag(Qh + (size_t)m0 * HD + 32, HD, lane);

  // all-ones bf16 B fragment for row sums
  FragCast onesc;
  {
    U4 u; u.x = u.y = u.z = u.w = 0x3F803F80u;  // bf16 1.0 pair
    onesc.q[0] = u; onesc.q[1] = u;
  }
  const v16bf onesb = onesc.v;

  const int rl = lane >> 4, cl = lane & 15;
  float mbias[8];
  #pragma unroll
  for (int r = 0; r < 8; ++r)
    mbias[r] = (mask[b * N + m0 + 8 * rl + r] == 0) ? -1.0e30f : 0.0f;

  const v8f vzero = {0.f, 0.f, 0.f, 0.f, 0.f, 0.f, 0.f, 0.f};
  v8f o[5];                                     // o[4] = running row-sum
  for (int i = 0; i < 5; ++i) o[i] = vzero;
  float mrun[8];
  #pragma unroll
  for (int r = 0; r < 8; ++r) mrun[r] = -3.0e38f;

  const float kS = 0.125f * 1.4426950408889634f;  // HD^-0.5 * log2(e)

  for (int kt = 0; kt < N; kt += 64) {
    v8f s[4];
    #pragma unroll
    for (int j = 0; j < 4; ++j) {
      s[j] = wmma_bf16(qf0, load_b_frag(Kh + (size_t)(kt + 16 * j) * HD + 0, HD, lane), vzero);
      s[j] = wmma_bf16(qf1, load_b_frag(Kh + (size_t)(kt + 16 * j) * HD + 32, HD, lane), s[j]);
    }

    float alpha[8];
    #pragma unroll
    for (int r = 0; r < 8; ++r) {
      const float a0 = fmaf(s[0][r], kS, mbias[r]);
      const float a1 = fmaf(s[1][r], kS, mbias[r]);
      const float a2 = fmaf(s[2][r], kS, mbias[r]);
      const float a3 = fmaf(s[3][r], kS, mbias[r]);
      float t = fmaxf(fmaxf(a0, a1), fmaxf(a2, a3));
      for (int off = 1; off < 16; off <<= 1)
        t = fmaxf(t, __shfl_xor(t, off, 32));
      const float mnew = fmaxf(mrun[r], t);
      alpha[r] = __builtin_amdgcn_exp2f(mrun[r] - mnew);
      mrun[r] = mnew;
      Ps[wave][8 * rl + r][cl +  0] = f2bf(__builtin_amdgcn_exp2f(a0 - mnew));
      Ps[wave][8 * rl + r][cl + 16] = f2bf(__builtin_amdgcn_exp2f(a1 - mnew));
      Ps[wave][8 * rl + r][cl + 32] = f2bf(__builtin_amdgcn_exp2f(a2 - mnew));
      Ps[wave][8 * rl + r][cl + 48] = f2bf(__builtin_amdgcn_exp2f(a3 - mnew));
    }
    #pragma unroll
    for (int i = 0; i < 5; ++i)
      #pragma unroll
      for (int r = 0; r < 8; ++r) o[i][r] *= alpha[r];

    asm volatile("" ::: "memory");  // keep LDS store->load order (DS in-order per wave)
    const v16bf pf0 = load_a_frag(&Ps[wave][0][0],  PLD, lane);  // keys kt..kt+31
    const v16bf pf1 = load_a_frag(&Ps[wave][0][32], PLD, lane);  // keys kt+32..kt+63
    #pragma unroll
    for (int dt = 0; dt < 4; ++dt) {
      o[dt] = wmma_bf16(pf0, load_b_frag(Vh + (size_t)dt * 16 * N + kt +  0, N, lane), o[dt]);
      o[dt] = wmma_bf16(pf1, load_b_frag(Vh + (size_t)dt * 16 * N + kt + 32, N, lane), o[dt]);
    }
    o[4] = wmma_bf16(pf0, onesb, o[4]);
    o[4] = wmma_bf16(pf1, onesb, o[4]);
  }

  #pragma unroll
  for (int r = 0; r < 8; ++r) {
    const float rinv = __builtin_amdgcn_rcpf(o[4][r]);
    const int row = m0 + 8 * rl + r;
    #pragma unroll
    for (int dt = 0; dt < 4; ++dt)
      vals[((size_t)b * N + row) * 1024 + h * 64 + dt * 16 + cl] =
          f2bf(o[dt][r] * rinv);
  }
}

// ---------------------------------------------------------------------------
extern "C" void kernel_launch(void* const* d_in, const int* in_sizes, int n_in,
                              void* d_out, int out_size, void* d_ws, size_t ws_size,
                              hipStream_t stream) {
  const float* x      = (const float*)d_in[0];
  const int*   mask   = (const int*)d_in[1];
  const float* qkv_w  = (const float*)d_in[2];
  const float* qkv_b  = (const float*)d_in[3];
  const float* proj_w = (const float*)d_in[4];
  const float* proj_b = (const float*)d_in[5];
  float* out = (float*)d_out;

  char* ws = (char*)d_ws;
  unsigned short* xb    = (unsigned short*)ws; ws += 4096ull * 1024 * 2;
  unsigned short* wqkv  = (unsigned short*)ws; ws += 3072ull * 1024 * 2;
  unsigned short* wproj = (unsigned short*)ws; ws += 1024ull * 1024 * 2;
  unsigned short* Qb    = (unsigned short*)ws; ws += 32ull * 2048 * 64 * 2;
  unsigned short* Kb    = (unsigned short*)ws; ws += 32ull * 2048 * 64 * 2;
  unsigned short* Vtb   = (unsigned short*)ws; ws += 32ull * 64 * 2048 * 2;
  unsigned short* vals  = (unsigned short*)ws; ws += 4096ull * 1024 * 2;
  (void)ws_size; (void)in_sizes; (void)n_in; (void)out_size;

  k_cvt_bf16<<<2048, 256, 0, stream>>>(x, xb, 4096 * 1024);
  k_cvt_bf16<<<2048, 256, 0, stream>>>(qkv_w, wqkv, 3072 * 1024);
  k_cvt_bf16<<<1024, 256, 0, stream>>>(proj_w, wproj, 1024 * 1024);

  k_gemm_bf16<0><<<dim3(32, 24), 256, 0, stream>>>(
      xb, wqkv, qkv_b, nullptr, Qb, Kb, Vtb, 4096, 3072, 1024);

  k_attn<<<1024, 128, 0, stream>>>(Qb, Kb, Vtb, mask, vals);

  k_gemm_bf16<1><<<dim3(32, 8), 256, 0, stream>>>(
      vals, wproj, proj_b, out, nullptr, nullptr, nullptr, 4096, 1024, 1024);
}
